// Attention_40346922778786
// MI455X (gfx1250) — matrix operand order repaired and checked by reading, not compiled, not run on hardware
//
#include <hip/hip_runtime.h>
#include <math.h>

// ---------------- problem constants ----------------
#define BB   64
#define CC   512
#define HH   28
#define WW   28
#define HWP  784            // 28*28
#define BC   32768          // B*C
#define XCNT 25690112       // B*C*H*W

// ---------------- workspace layout (float elements) ----------------
// [0, XCNT)           : x ping buffer
// then small f32 state, then an f16 region.
constexpr int SEQ_OFF  = XCNT;
constexpr int HT_OFF   = SEQ_OFF  + BC;
constexpr int CT_OFF   = HT_OFF   + BC;
constexpr int OUTH_OFF = CT_OFF   + BC;
constexpr int GI_OFF   = OUTH_OFF + BC;        // 64*1536 floats
constexpr int FEND     = GI_OFF   + 64 * 1536;

// f16 region element offsets (base = (_Float16*)(ws + FEND))
constexpr int SEQH_OFF = 0;
constexpr int HTH_OFF  = SEQH_OFF + BC;
constexpr int W1IH_OFF = HTH_OFF  + BC;        // 32*512
constexpr int W1HH_OFF = W1IH_OFF + 16384;
constexpr int W2IH_OFF = W1HH_OFF + 16384;     // 1536*32
constexpr int W2HH_OFF = W2IH_OFF + 49152;
constexpr int H1I_OFF  = W2HH_OFF + 49152;     // 64*32
constexpr int H1H_OFF  = H1I_OFF  + 2048;

typedef __attribute__((ext_vector_type(16))) _Float16 v16h;
typedef __attribute__((ext_vector_type(8)))  float    v8f;

__device__ inline _Float16* halfbase(float* ws) { return (_Float16*)(ws + FEND); }
__device__ inline const _Float16* halfbase(const float* ws) { return (const _Float16*)(ws + FEND); }

__device__ inline float sigf(float x) { return 1.0f / (1.0f + __expf(-x)); }

// ---------------- CDNA5 async global<->LDS plumbing ----------------
#if __has_builtin(__builtin_amdgcn_global_load_async_to_lds_b128) && \
    __has_builtin(__builtin_amdgcn_global_store_async_from_lds_b128)
#define USE_ASYNC_LDS 1
#else
#define USE_ASYNC_LDS 0
#endif

// Builtin params are GCC-vector int4 pointers: AS1 (prints as __device__) and
// AS3 (prints as __shared__) per the clang diagnostics.
typedef int v4i_g __attribute__((vector_size(16)));
typedef __attribute__((address_space(1))) v4i_g gv4i;
typedef __attribute__((address_space(3))) v4i_g lv4i;

__device__ inline gv4i* asg(const void* p) {
  return (gv4i*)(unsigned long long)(uintptr_t)p;
}
__device__ inline lv4i* asl(const void* p) {
  // generic LDS address: low 32 bits are the wave-relative LDS byte address
  return (lv4i*)(unsigned int)(uintptr_t)p;
}
__device__ inline void wait_async0() {
#if __has_builtin(__builtin_amdgcn_s_wait_asynccnt)
  __builtin_amdgcn_s_wait_asynccnt(0);
#else
  asm volatile("s_wait_asynccnt 0" ::: "memory");
#endif
}

// D = A*B + C, 16x16x32 f16->f32 (CDNA5 WMMA)
__device__ inline v8f wmma16(v16h a, v16h b, v8f c) {
  return __builtin_amdgcn_wmma_f32_16x16x32_f16(false, a, false, b, (short)0, c, false, false);
}

// A operand gather: A row-major [M x lda] f16. ISA 16-bit A 16x32 layout:
// lane = half*16 + row ; element i in 0..7  -> K = kbase + i   + 8*half
//                        element i in 8..15 -> K = kbase + i+8 + 8*half
__device__ inline v16h load_A(const _Float16* __restrict__ A, int lda, int mbase,
                              int kbase, int lane) {
  const int row = lane & 15, hf = lane >> 4;
  v16h a;
#pragma unroll
  for (int i = 0; i < 16; ++i) {
    const int k = kbase + (i & 7) + 8 * hf + ((i >> 3) << 4);
    a[i] = A[(mbase + row) * lda + k];
  }
  return a;
}

// B operand gather: B[k][n] = Wt[n][k], Wt row-major [N x ldb] f16.
// ISA 16-bit B 32x16 layout: N = lane&15, element i -> K = kbase + (i&7) + 8*half + 16*(i>>3).
__device__ inline v16h load_B(const _Float16* __restrict__ Wt, int ldb, int nbase,
                              int kbase, int lane) {
  const int col = lane & 15, hf = lane >> 4;
  v16h b;
#pragma unroll
  for (int i = 0; i < 16; ++i) {
    const int k = kbase + (i & 7) + 8 * hf + ((i >> 3) << 4);
    b[i] = Wt[(nbase + col) * ldb + k];
  }
  return b;
}

// ---------------- prep: f16 weight conversion + zero state ----------------
__global__ __launch_bounds__(256) void prep_kernel(
    const float* __restrict__ w1ih, const float* __restrict__ w1hh,
    const float* __restrict__ w2ih, const float* __restrict__ w2hh,
    float* __restrict__ ws) {
  const int i = blockIdx.x * 256 + threadIdx.x;
  _Float16* hb = halfbase(ws);
  if (i < 16384) {
    hb[W1IH_OFF + i] = (_Float16)w1ih[i];
  } else if (i < 32768) {
    hb[W1HH_OFF + (i - 16384)] = (_Float16)w1hh[i - 16384];
  } else if (i < 81920) {
    hb[W2IH_OFF + (i - 32768)] = (_Float16)w2ih[i - 32768];
  } else if (i < 131072) {
    hb[W2HH_OFF + (i - 81920)] = (_Float16)w2hh[i - 81920];
  } else if (i < 163840) {
    const int j = i - 131072;
    ws[HT_OFF + j] = 0.0f;
    ws[CT_OFF + j] = 0.0f;
    hb[HTH_OFF + j] = (_Float16)0.0f;
  }
}

// ---------------- layer-0 pooling: one wave per (b,c) plane ----------------
__global__ __launch_bounds__(256) void pool0_kernel(const float* __restrict__ x,
                                                    float* __restrict__ ws) {
  const int plane = blockIdx.x * 8 + (threadIdx.x >> 5);
  const int lane  = threadIdx.x & 31;
  const float4* p4 = (const float4*)(x + (size_t)plane * HWP);
  float s = 0.0f;
  for (int i = lane; i < HWP / 4; i += 32) {   // 196 float4 per plane
    const float4 v = p4[i];
    s += v.x + v.y + v.z + v.w;
  }
#pragma unroll
  for (int off = 16; off; off >>= 1) s += __shfl_down(s, off, 32);
  if (lane == 0) {
    const float m = s * (1.0f / (float)HWP);
    ws[SEQ_OFF + plane] = m;
    halfbase(ws)[SEQH_OFF + plane] = (_Float16)m;
  }
}

// ---------------- cell stage 1: H1 = relu(X @ W1^T + b), [64,32] x2 -------
// 16 output tiles (ih:8 + hh:8), 8 waves, 2 tiles/wave, K-loop 512/32.
__global__ __launch_bounds__(256) void cell_l1_kernel(
    const float* __restrict__ b_ih_l1, const float* __restrict__ b_hh_l1,
    float* __restrict__ ws) {
  const int wav = threadIdx.x >> 5, lane = threadIdx.x & 31;
  const int col = lane & 15, hf = lane >> 4;
  const _Float16* hb = halfbase(ws);
  _Float16* hbw = halfbase(ws);
#pragma unroll
  for (int tt = 0; tt < 2; ++tt) {
    const int t = wav * 2 + tt;         // 0..15
    const int which = t >> 3;           // 0: ih path, 1: hh path
    const int rem = t & 7;
    const int mtile = rem >> 1, ntile = rem & 1;
    const _Float16* Ah = hb + (which ? HTH_OFF : SEQH_OFF);
    const _Float16* Wh = hb + (which ? W1HH_OFF : W1IH_OFF);
    const float* bias  = which ? b_hh_l1 : b_ih_l1;
    _Float16* H1 = hbw + (which ? H1H_OFF : H1I_OFF);
    v8f acc = {};
    for (int kk = 0; kk < CC; kk += 32) {
      v16h a = load_A(Ah, CC, mtile * 16, kk, lane);
      v16h b = load_B(Wh, CC, ntile * 16, kk, lane);
      acc = wmma16(a, b, acc);
    }
    const int n = ntile * 16 + col;
    const float bv = bias[n];
#pragma unroll
    for (int r = 0; r < 8; ++r) {
      const int m = mtile * 16 + r + 8 * hf;
      const float v = acc[r] + bv;
      H1[m * 32 + n] = (_Float16)(v > 0.0f ? v : 0.0f);
    }
  }
}

// ------- cell stage 2: G = H1i@W2ih^T + H1h@W2hh^T (+biases); LSTM update --
// Gate columns couple (c, c+512, c+1024), so each wave owns (mtile, jc) pairs
// and computes the 3 N-tiles of that gate triple. Also caches Gi (w/ bias)
// for the second cell, which reuses the identical g_i.
__global__ __launch_bounds__(256) void cell_l2_kernel(
    const float* __restrict__ b_ih_l2, const float* __restrict__ b_hh_l2,
    float* __restrict__ ws) {
  const int gw = blockIdx.x * 8 + (threadIdx.x >> 5);  // 0..63
  const int lane = threadIdx.x & 31;
  const int col = lane & 15, hf = lane >> 4;
  const _Float16* hb = halfbase(ws);
  const _Float16* h1i  = hb + H1I_OFF;
  const _Float16* h1h  = hb + H1H_OFF;
  const _Float16* w2ih = hb + W2IH_OFF;
  const _Float16* w2hh = hb + W2HH_OFF;
  float* gi   = ws + GI_OFF;
  float* htf  = ws + HT_OFF;
  float* ctf  = ws + CT_OFF;
  _Float16* hth = halfbase(ws) + HTH_OFF;
#pragma unroll
  for (int pp = 0; pp < 2; ++pp) {
    const int p = gw * 2 + pp;          // 0..127
    const int mtile = p >> 5;           // 0..3 (batch tiles)
    const int jc = p & 31;              // 0..31 (c column tiles)
    v16h ai = load_A(h1i, 32, mtile * 16, 0, lane);
    v16h ah = load_A(h1h, 32, mtile * 16, 0, lane);
    v8f giv[3], gsum[3];
#pragma unroll
    for (int g = 0; g < 3; ++g) {
      const int n0 = jc * 16 + 512 * g;
      v8f z0 = {};
      v16h bi = load_B(w2ih, 32, n0, 0, lane);
      v8f gi_t = wmma16(ai, bi, z0);
      v8f z1 = {};
      v16h bh = load_B(w2hh, 32, n0, 0, lane);
      v8f gh_t = wmma16(ah, bh, z1);
      const int n = n0 + col;
      const float bivv = b_ih_l2[n], bhvv = b_hh_l2[n];
#pragma unroll
      for (int r = 0; r < 8; ++r) {
        giv[g][r]  = gi_t[r] + bivv;
        gsum[g][r] = giv[g][r] + gh_t[r] + bhvv;
      }
    }
#pragma unroll
    for (int r = 0; r < 8; ++r) {
      const int m = mtile * 16 + r + 8 * hf;
      const int c = jc * 16 + col;
#pragma unroll
      for (int g = 0; g < 3; ++g) gi[m * 1536 + g * 512 + c] = giv[g][r];
      const float ig = gsum[0][r], fg = gsum[1][r], cg = gsum[2][r];
      const float ncx = sigf(fg) * ctf[m * CC + c] + sigf(ig) * tanhf(cg);
      const float h = sigf(ncx);
      ctf[m * CC + c] = ncx;
      htf[m * CC + c] = h;
      hth[m * CC + c] = (_Float16)h;
    }
  }
}

// ------- cell 2 vector path: g_h(ht[0]) broadcast, then out_h [64,512] -----
__global__ __launch_bounds__(512) void cell_vec_kernel(
    const float* __restrict__ w_hh_l1, const float* __restrict__ b_hh_l1,
    const float* __restrict__ w_hh_l2, const float* __restrict__ b_hh_l2,
    float* __restrict__ ws) {
  __shared__ float tbuf[32];
  __shared__ float gh2s[1536];
  const int tid = threadIdx.x;
  const float* htf = ws + HT_OFF;   // row 0 = ht[0]
  const float* ctf = ws + CT_OFF;   // row 0 = ct[0]
  const float* gi  = ws + GI_OFF;
  float* outh = ws + OUTH_OFF;
  if (tid < 32) {
    float s = b_hh_l1[tid];
    for (int k = 0; k < CC; ++k) s += htf[k] * w_hh_l1[tid * CC + k];
    tbuf[tid] = s > 0.0f ? s : 0.0f;
  }
  __syncthreads();
  for (int n = tid; n < 1536; n += 512) {
    float s = b_hh_l2[n];
#pragma unroll
    for (int k = 0; k < 32; ++k) s += tbuf[k] * w_hh_l2[n * 32 + k];
    gh2s[n] = s;
  }
  __syncthreads();
  for (int idx = tid; idx < BC; idx += 512) {
    const int b = idx >> 9, c = idx & 511;
    const float i2 = gi[b * 1536 + c]        + gh2s[c];
    const float f2 = gi[b * 1536 + 512 + c]  + gh2s[512 + c];
    const float c2 = gi[b * 1536 + 1024 + c] + gh2s[1024 + c];
    const float ncx = sigf(f2) * ctf[c] + sigf(i2) * tanhf(c2);
    outh[idx] = sigf(ncx);
  }
}

// ------- fused depthwise 3x3 + scale + residual + next-layer pooling -------
// Plane (784 f32 = 3136B, 16B aligned) staged via CDNA5 async global<->LDS.
__global__ __launch_bounds__(256) void conv_fuse_kernel(
    const float* __restrict__ xin, float* __restrict__ xout,
    const float* __restrict__ dwk, float* __restrict__ ws) {
  const int bc = blockIdx.x;
  const int c = bc & (CC - 1);
  const size_t plane = (size_t)bc * HWP;
  __shared__ __align__(16) float tile[HWP];
  __shared__ float wsum[8];
#if USE_ASYNC_LDS
  __shared__ __align__(16) float tile2[HWP];
#endif
  const int tid = threadIdx.x;

#if USE_ASYNC_LDS
  if (tid < HWP / 4) {   // 196 x 16B async copies, no VGPR round-trip
    __builtin_amdgcn_global_load_async_to_lds_b128(
        asg(xin + plane + (size_t)tid * 4), asl(&tile[tid * 4]), 0, 0);
  }
#else
  if (tid < HWP / 4) {
    const float4 v = ((const float4*)(xin + plane))[tid];
    ((float4*)tile)[tid] = v;
  }
#endif
  const float* kc = dwk + c * 9;
  const float k0 = kc[0], k1 = kc[1], k2 = kc[2];
  const float k3 = kc[3], k4 = kc[4], k5 = kc[5];
  const float k6 = kc[6], k7 = kc[7], k8 = kc[8];
  const float s = 1.0f + ws[OUTH_OFF + bc];
#if USE_ASYNC_LDS
  wait_async0();
#endif
  __syncthreads();

  float lsum = 0.0f;
  for (int i = tid; i < HWP; i += 256) {
    const int h = i / WW, w = i - h * WW;
    auto T = [&](int hh, int ww) -> float {
      return (hh >= 0 && hh < HH && ww >= 0 && ww < WW) ? tile[hh * WW + ww] : 0.0f;
    };
    const float conv =
        k0 * T(h - 1, w - 1) + k1 * T(h - 1, w) + k2 * T(h - 1, w + 1) +
        k3 * T(h,     w - 1) + k4 * tile[i]     + k5 * T(h,     w + 1) +
        k6 * T(h + 1, w - 1) + k7 * T(h + 1, w) + k8 * T(h + 1, w + 1);
    const float v = tile[i] * s + conv;  // x*(1+out_h) + dwconv(x)
#if USE_ASYNC_LDS
    tile2[i] = v;
#else
    xout[plane + i] = v;
#endif
    lsum += v;
  }
#pragma unroll
  for (int off = 16; off; off >>= 1) lsum += __shfl_down(lsum, off, 32);
  if ((tid & 31) == 0) wsum[tid >> 5] = lsum;
  __syncthreads();   // tile2 complete + wsum visible

#if USE_ASYNC_LDS
  if (tid < HWP / 4) {   // async store LDS -> global (drained at ENDPGM)
    __builtin_amdgcn_global_store_async_from_lds_b128(
        asg(xout + plane + (size_t)tid * 4), asl(&tile2[tid * 4]), 0, 0);
  }
#endif
  if (tid == 0) {
    float t = 0.0f;
#pragma unroll
    for (int i = 0; i < 8; ++i) t += wsum[i];
    const float m = t * (1.0f / (float)HWP);
    ws[SEQ_OFF + bc] = m;                       // next layer's pooled seq
    halfbase(ws)[SEQH_OFF + bc] = (_Float16)m;
  }
}

extern "C" void kernel_launch(void* const* d_in, const int* in_sizes, int n_in,
                              void* d_out, int out_size, void* d_ws, size_t ws_size,
                              hipStream_t stream) {
  const float* x        = (const float*)d_in[0];
  const float* w_ih_l1  = (const float*)d_in[1];
  const float* b_ih_l1  = (const float*)d_in[2];
  const float* w_ih_l2  = (const float*)d_in[3];
  const float* b_ih_l2  = (const float*)d_in[4];
  const float* w_hh_l1  = (const float*)d_in[5];
  const float* b_hh_l1  = (const float*)d_in[6];
  const float* w_hh_l2  = (const float*)d_in[7];
  const float* b_hh_l2  = (const float*)d_in[8];
  const float* dwk      = (const float*)d_in[9];
  float* out = (float*)d_out;
  float* ws  = (float*)d_ws;

  prep_kernel<<<640, 256, 0, stream>>>(w_ih_l1, w_hh_l1, w_ih_l2, w_hh_l2, ws);
  pool0_kernel<<<BC / 8, 256, 0, stream>>>(x, ws);

  const float* cur = x;
  float* dsts[4] = { ws, out, ws, out };   // ping-pong, ends in d_out
  for (int l = 0; l < 4; ++l) {            // num_layers == 4
    cell_l1_kernel<<<1, 256, 0, stream>>>(b_ih_l1, b_hh_l1, ws);
    cell_l2_kernel<<<8, 256, 0, stream>>>(b_ih_l2, b_hh_l2, ws);
    cell_vec_kernel<<<1, 512, 0, stream>>>(w_hh_l1, b_hh_l1, w_hh_l2, b_hh_l2, ws);
    conv_fuse_kernel<<<BC, 256, 0, stream>>>(cur, dsts[l], dwk, ws);
    cur = dsts[l];
  }
}
